// GCN_58162447123289
// MI455X (gfx1250) — matrix operand order repaired and checked by reading, not compiled, not run on hardware
//
#include <hip/hip_runtime.h>

typedef __attribute__((ext_vector_type(16))) _Float16 v16h;
typedef __attribute__((ext_vector_type(8)))  _Float16 v8h;
typedef __attribute__((ext_vector_type(4)))  _Float16 v4h;
typedef __attribute__((ext_vector_type(8)))  float    v8f;

#define GCN_N   100000
#define GCN_E   3200000
#define GCN_F   512
#define GCN_H   256

#define LDS_STRIDE 40   // f16 elems; 80B row stride: 16B-aligned, conflict-free

// Assemble a 16-element f16 fragment from two 16B LDS chunks (2x ds_load_b128)
__device__ __forceinline__ v16h gcn_frag16(const _Float16* p0, const _Float16* p1) {
  const v8h lo = *(const v8h*)p0;
  const v8h hi = *(const v8h*)p1;
  v16h r;
#pragma unroll
  for (int i = 0; i < 8; ++i) { r[i] = lo[i]; r[8 + i] = hi[i]; }
  return r;
}

// One-time pack: Wt[n][k] = (f16)W[k][n].  W is 0.25-0.5 MB and re-read by
// 3125 blocks, so pre-transposing/converting removes all strided loads and
// cvt chains from the GEMM hot loop.
__global__ __launch_bounds__(256) void gcn_pack_wt(
    const float* __restrict__ W, _Float16* __restrict__ Wt, int K) {
  const int kq4 = K >> 2;
  const int idx = blockIdx.x * blockDim.x + threadIdx.x;
  if (idx >= GCN_H * kq4) return;
  const int n = idx / kq4;
  const int k = (idx - n * kq4) * 4;
  v4h h;
#pragma unroll
  for (int i = 0; i < 4; ++i)
    h[i] = (_Float16)W[(size_t)(k + i) * GCN_H + n];
  *(v4h*)&Wt[(size_t)n * K + k] = h;
}

// ---------------------------------------------------------------------------
// C[M x 256] = A[M x K] * Bt^T; Bt is f16 pre-transposed [256][K];
// A is fp32 (AHALF=false) or f16 (AHALF=true).
// Block = 256 threads (8 waves) -> 32 x 256 output tile; one wave = 16 x 64.
// Double-buffered LDS pipeline: stage slab k+32 while computing slab k.
//
// Fragment layouts (CDNA5 ISA 7.12.2, wave32), khalf = lane>>4:
//   A elems 0..7  -> k = 8*khalf + i ; elems 8..15 -> k = 16 + 8*khalf + i
//   B elems 0..7  -> k = 8*khalf + i ; elems 8..15 -> k = 16 + 8*khalf + i  (lane&15 = N column)
//   C elem r      -> M = 8*khalf + r, N = lane&15
// ---------------------------------------------------------------------------
template <bool AHALF>
__global__ __launch_bounds__(256) void gcn_gemm_wmma(
    const void* __restrict__ Av, const _Float16* __restrict__ Bt,
    float* __restrict__ C, int K) {
  __shared__ __align__(16) _Float16 Alds[2][32 * LDS_STRIDE];
  __shared__ __align__(16) _Float16 Blds[2][256 * LDS_STRIDE];

  const int tid   = threadIdx.x;
  const int lane  = tid & 31;
  const int wav   = tid >> 5;
  const int wm    = wav & 1;        // 16-row sub-tile
  const int wn    = wav >> 1;       // 64-col strip (4 WMMA tiles)
  const int khalf = lane >> 4;
  const int l16   = lane & 15;
  const size_t rowbase = (size_t)blockIdx.x * 32;

  const int ar = tid >> 3;          // A staging: row 0..31
  const int ak = (tid & 7) * 4;     //            k quad
  const int bn = tid;               // B staging: one of 256 columns

  const float*    Af = (const float*)Av;
  const _Float16* Ah = (const _Float16*)Av;

  // ---- stage one k-slab into LDS buffer `buf` ----
  auto stage = [&](int kk, int buf) {
    if constexpr (AHALF) {
      const v4h av = *(const v4h*)(Ah + (rowbase + ar) * K + (kk + ak));
      *(v4h*)&Alds[buf][ar * LDS_STRIDE + ak] = av;
    } else {
      const float4 af = *(const float4*)(Af + (rowbase + ar) * K + (kk + ak));
      v4h h;
      h[0] = (_Float16)af.x; h[1] = (_Float16)af.y;
      h[2] = (_Float16)af.z; h[3] = (_Float16)af.w;
      *(v4h*)&Alds[buf][ar * LDS_STRIDE + ak] = h;
    }
    // B already f16 + transposed: 4 contiguous b128 copies, no cvt
#pragma unroll
    for (int c8 = 0; c8 < 4; ++c8) {
      const v8h p = *(const v8h*)&Bt[(size_t)bn * K + kk + c8 * 8];
      *(v8h*)&Blds[buf][bn * LDS_STRIDE + c8 * 8] = p;
    }
  };

  v8f acc[4] = {};
  stage(0, 0);

  int cur = 0;
  for (int k0 = 0; k0 < K; k0 += 32, cur ^= 1) {
    __syncthreads();   // staging of `cur` visible; prior reads of cur^1 done

    // ---- pre-load ALL fragments for this slab (waits hoisted, WMMAs b2b) ----
    const _Float16* abase = &Alds[cur][(wm * 16 + l16) * LDS_STRIDE + khalf * 8];
    const v16h a = gcn_frag16(abase, abase + 16);
    v16h bfr[4];
#pragma unroll
    for (int j = 0; j < 4; ++j) {
      const _Float16* bbase =
          &Blds[cur][(wn * 64 + j * 16 + l16) * LDS_STRIDE + khalf * 8];
      bfr[j] = gcn_frag16(bbase, bbase + 16);
    }

    // ---- stage next slab into other buffer (overlaps with WMMAs) ----
    if (k0 + 32 < K) {
      stage(k0 + 32, cur ^ 1);
      if (k0 + 64 < K) {   // two-ahead L2 prefetch (global_prefetch_b8)
        __builtin_prefetch(&Bt[(size_t)bn * K + k0 + 64], 0, 3);
        if constexpr (AHALF)
          __builtin_prefetch(Ah + (rowbase + ar) * K + (k0 + 64 + ak), 0, 3);
        else
          __builtin_prefetch(Af + (rowbase + ar) * K + (k0 + 64 + ak), 0, 3);
      }
    }

    // ---- 4 back-to-back WMMAs sharing one A fragment ----
#pragma unroll
    for (int j = 0; j < 4; ++j)
      acc[j] = __builtin_amdgcn_wmma_f32_16x16x32_f16(
          /*neg_a=*/false, a, /*neg_b=*/false, bfr[j],
          /*c_mod=*/(short)0, acc[j], /*reuse_a=*/false, /*reuse_b=*/false);
  }

  const size_t row0 = rowbase + wm * 16 + khalf * 8;
#pragma unroll
  for (int j = 0; j < 4; ++j) {
    float* cp = C + row0 * GCN_H + (wn * 64 + j * 16 + l16);
#pragma unroll
    for (int r = 0; r < 8; ++r) cp[(size_t)r * GCN_H] = acc[j][r];
  }
}

// ---------------------------------------------------------------------------
// SPMM scatter: out[row[e]] += val[e] * S[col[e]], one wave32 per edge.
// Gather (102 MB) and scatter (102 MB) targets are L2-resident (192 MB L2).
// Relaxed agent-scope fetch_add -> non-returning global_atomic_add_f32.
// ---------------------------------------------------------------------------
__global__ __launch_bounds__(256) void gcn_spmm_edges(
    const int* __restrict__ erow, const int* __restrict__ ecol,
    const float* __restrict__ eval, const float* __restrict__ S,
    float* __restrict__ out, int E) {
  const int wave = blockIdx.x * (blockDim.x >> 5) + (threadIdx.x >> 5);
  const int lane = threadIdx.x & 31;
  if (wave >= E) return;
  const int r   = erow[wave];
  const int c   = ecol[wave];
  const float v = eval[wave];
  const float4* __restrict__ src = (const float4*)(S + (size_t)c * GCN_H);
  float* __restrict__ dst        = out + (size_t)r * GCN_H;
#pragma unroll
  for (int i = 0; i < 2; ++i) {
    const int q = lane + 32 * i;          // float4 index 0..63
    const float4 s4 = src[q];
    float* d = dst + q * 4;
    __hip_atomic_fetch_add(d + 0, v * s4.x, __ATOMIC_RELAXED, __HIP_MEMORY_SCOPE_AGENT);
    __hip_atomic_fetch_add(d + 1, v * s4.y, __ATOMIC_RELAXED, __HIP_MEMORY_SCOPE_AGENT);
    __hip_atomic_fetch_add(d + 2, v * s4.z, __ATOMIC_RELAXED, __HIP_MEMORY_SCOPE_AGENT);
    __hip_atomic_fetch_add(d + 3, v * s4.w, __ATOMIC_RELAXED, __HIP_MEMORY_SCOPE_AGENT);
  }
}

// out_f16 = prelu(y + bias); h1 is consumed only as the f16 A of GEMM2,
// so emit f16 directly (halves write + reread traffic). 4-wide.
__global__ __launch_bounds__(256) void gcn_bias_prelu_f16(
    const float* __restrict__ y, const float* __restrict__ bias,
    const float* __restrict__ alpha, _Float16* __restrict__ out, int total4) {
  const int i = blockIdx.x * blockDim.x + threadIdx.x;
  if (i >= total4) return;
  const float a  = alpha[0];
  const float4 v4 = ((const float4*)y)[i];
  const float4 b4 = *(const float4*)&bias[(i * 4) & (GCN_H - 1)];
  v4h o;
  float t;
  t = v4.x + b4.x; o[0] = (_Float16)((t >= 0.f) ? t : a * t);
  t = v4.y + b4.y; o[1] = (_Float16)((t >= 0.f) ? t : a * t);
  t = v4.z + b4.z; o[2] = (_Float16)((t >= 0.f) ? t : a * t);
  t = v4.w + b4.w; o[3] = (_Float16)((t >= 0.f) ? t : a * t);
  *(v4h*)&out[i * 4] = o;
}

// y = prelu(y + bias) in place (final f32 output), 4-wide
__global__ __launch_bounds__(256) void gcn_bias_prelu_f32(
    float* __restrict__ y, const float* __restrict__ bias,
    const float* __restrict__ alpha, int total4) {
  const int i = blockIdx.x * blockDim.x + threadIdx.x;
  if (i >= total4) return;
  const float a  = alpha[0];
  float4 v4 = ((const float4*)y)[i];
  const float4 b4 = *(const float4*)&bias[(i * 4) & (GCN_H - 1)];
  v4.x += b4.x; v4.x = (v4.x >= 0.f) ? v4.x : a * v4.x;
  v4.y += b4.y; v4.y = (v4.y >= 0.f) ? v4.y : a * v4.y;
  v4.z += b4.z; v4.z = (v4.z >= 0.f) ? v4.z : a * v4.z;
  v4.w += b4.w; v4.w = (v4.w >= 0.f) ? v4.w : a * v4.w;
  ((float4*)y)[i] = v4;
}

__global__ __launch_bounds__(256) void gcn_fill_zero(float4* __restrict__ p, int n4) {
  const int i = blockIdx.x * blockDim.x + threadIdx.x;
  if (i < n4) p[i] = make_float4(0.f, 0.f, 0.f, 0.f);
}

// ---------------------------------------------------------------------------
extern "C" void kernel_launch(void* const* d_in, const int* in_sizes, int n_in,
                              void* d_out, int out_size, void* d_ws, size_t ws_size,
                              hipStream_t stream) {
  const float* x     = (const float*)d_in[0];
  const int*   erow  = (const int*)  d_in[1];
  const int*   ecol  = (const int*)  d_in[2];
  const float* eval_ = (const float*)d_in[3];
  const float* W1    = (const float*)d_in[4];
  const float* b1    = (const float*)d_in[5];
  const float* W2    = (const float*)d_in[6];
  const float* b2    = (const float*)d_in[7];
  const float* alpha = (const float*)d_in[8];
  float* out = (float*)d_out;

  float*     bufA = (float*)d_ws;                    // [N,256] f32: s1, then h1(f16) alias
  float*     bufB = bufA + (size_t)GCN_N * GCN_H;    // [N,256] f32: acc1, then s2
  _Float16*  bufH = (_Float16*)bufA;                 // h1 in f16 (s1 dead by then)
  _Float16*  Wt1  = (_Float16*)(bufB + (size_t)GCN_N * GCN_H);   // [256][512] f16
  _Float16*  Wt2  = Wt1 + (size_t)GCN_H * GCN_F;                 // [256][256] f16

  const dim3 blk(256);
  const int elems       = GCN_N * GCN_H;             // 25.6M
  const int gemm_blocks = GCN_N / 32;                // 3125, exact
  const int spmm_blocks = (GCN_E + 7) / 8;           // 8 edge-waves per block
  const int q4_blocks   = (elems / 4 + 255) / 256;
  const int pk1_blocks  = (GCN_H * (GCN_F / 4) + 255) / 256;
  const int pk2_blocks  = (GCN_H * (GCN_H / 4) + 255) / 256;

  // One-time weight pack (transpose + f16)
  gcn_pack_wt<<<pk1_blocks, blk, 0, stream>>>(W1, Wt1, GCN_F);
  gcn_pack_wt<<<pk2_blocks, blk, 0, stream>>>(W2, Wt2, GCN_H);

  // Layer 1: s1 = x@W1 ; h1 = prelu(spmm(s1) + b1)  (h1 stored f16)
  gcn_gemm_wmma<false><<<gemm_blocks, blk, 0, stream>>>(x, Wt1, bufA, GCN_F);
  gcn_fill_zero<<<q4_blocks, blk, 0, stream>>>((float4*)bufB, elems / 4);
  gcn_spmm_edges<<<spmm_blocks, blk, 0, stream>>>(erow, ecol, eval_, bufA, bufB, GCN_E);
  gcn_bias_prelu_f16<<<q4_blocks, blk, 0, stream>>>(bufB, b1, alpha, bufH, elems / 4);

  // Layer 2: s2 = h1@W2 ; out = prelu(spmm(s2) + b2)
  gcn_gemm_wmma<true><<<gemm_blocks, blk, 0, stream>>>(bufH, Wt2, bufB, GCN_H);
  gcn_fill_zero<<<q4_blocks, blk, 0, stream>>>((float4*)out, elems / 4);
  gcn_spmm_edges<<<spmm_blocks, blk, 0, stream>>>(erow, ecol, eval_, bufB, out, GCN_E);
  gcn_bias_prelu_f32<<<q4_blocks, blk, 0, stream>>>(out, b2, alpha, elems / 4);
}
